// STHDformer_80590766342719
// MI455X (gfx1250) — compile-verified, matrix operand order checked by reading, not run on hardware
//
#include <hip/hip_runtime.h>

typedef __attribute__((ext_vector_type(16))) _Float16 v16h;
typedef __attribute__((ext_vector_type(8)))  _Float16 v8h;
typedef __attribute__((ext_vector_type(8)))  float    v8f;

// ---- model dims ----
#define BB      32
#define TT      12
#define NNODES  716
#define DD      152
#define DPAD    160          // D padded to multiple of 32 for WMMA K
#define LDP     168          // LDS f16 row stride (pads banks; 336B, 16B aligned)
#define NHEAD   4
#define HDIM    38
#define FFD     512
#define FFC     128          // FFN chunk width
#define LDH     136          // hidden-chunk LDS stride (272B, 16B aligned)
#define OUTS    12
#define SPB     8            // sequences per block
#define MROWS   96           // SPB*TT, = 6 * 16 (exact WMMA M tiles)
#define MT      6
#define NT_D    10           // DPAD/16
#define KT_D    5            // DPAD/32
#define NTHREADS 256
#define NWAVES   8
#define BUFH    (MROWS*LDP)          // 16128 halfs per activation buffer
#define SMEM_BYTES (5*BUFH*2)        // 161280 B -> 2 blocks per WGP

// ---- f16 transposed-weight workspace layout (element offsets) ----
#define WQT_OFF 0
#define WKT_OFF (DPAD*DPAD)
#define WVT_OFF (2*DPAD*DPAD)
#define WOT_OFF (3*DPAD*DPAD)
#define W1T_OFF (4*DPAD*DPAD)              // FFD x DPAD
#define W2T_OFF (4*DPAD*DPAD + FFD*DPAD)   // DPAD x FFD
#define LAYER_WS (4*DPAD*DPAD + 2*FFD*DPAD)

struct LayerP {
  const float *bq,*bk,*bv,*bo,*b1,*b2,*g1,*be1,*g2,*be2;
};
struct Params {
  const float *x,*Wi,*bi,*todT,*dowT,*adp,*Wout,*bout;
  const _Float16* W;   // pre-transposed f16 weights (d_ws)
  float* out;
  LayerP L[4];
};

// ---- WMMA fragment load: identical per-lane pattern for A (row-major act)
// and B (row-major W^T), matching the CDNA5 16-bit 16x32 A layout. ----
__device__ __forceinline__ v16h load_frag(const _Float16* base, int r0, int k0,
                                          int ld, int lane) {
  const int m   = r0 + (lane & 15);
  const int off = m * ld + k0 + ((lane & 16) ? 8 : 0);
  union { v16h v; v8h h[2]; } u;
  u.h[0] = *(const v8h*)(base + off);        // K = k0 .. k0+7   (lane-half adj)
  u.h[1] = *(const v8h*)(base + off + 16);   // K = k0+16 .. k0+23
  return u.v;
}

// C = A(LDS) @ W^T(global) + bias, optional relu, f16 store to LDS
__device__ __forceinline__ void gemm_f16out(
    const _Float16* __restrict__ A, int lda,
    const _Float16* __restrict__ Wt, int ldw,
    const float* __restrict__ bias, int nbias,
    _Float16* __restrict__ O, int ldo,
    int mtiles, int ntiles, int ktiles, bool relu, int wid, int lane)
{
  const int tiles = mtiles * ntiles;
  for (int t = wid; t < tiles; t += NWAVES) {
    const int r0 = (t / ntiles) * 16;
    const int n0 = (t % ntiles) * 16;
    v8f acc = {0.f,0.f,0.f,0.f,0.f,0.f,0.f,0.f};
    for (int kt = 0; kt < ktiles; ++kt) {
      v16h a = load_frag(A,  r0, kt*32, lda, lane);
      v16h b = load_frag(Wt, n0, kt*32, ldw, lane);
      acc = __builtin_amdgcn_wmma_f32_16x16x32_f16(false, a, false, b,
                                                   (short)0, acc, false, false);
    }
    const int n  = n0 + (lane & 15);
    const int mh = (lane & 16) ? 8 : 0;
    const float bv = (n < nbias) ? bias[n] : 0.f;
#pragma unroll
    for (int j = 0; j < 8; ++j) {
      float v = acc[j] + bv;
      if (relu) v = fmaxf(v, 0.f);
      O[(r0 + j + mh) * ldo + n] = (_Float16)v;
    }
  }
}

// Y(f32, LDS) += A(LDS f16) @ W^T(global f16)
__device__ __forceinline__ void gemm_f32acc(
    const _Float16* __restrict__ A, int lda,
    const _Float16* __restrict__ Wt, int ldw,
    float* __restrict__ Y, int ldy,
    int mtiles, int ntiles, int ktiles, int wid, int lane)
{
  const int tiles = mtiles * ntiles;
  for (int t = wid; t < tiles; t += NWAVES) {
    const int r0 = (t / ntiles) * 16;
    const int n0 = (t % ntiles) * 16;
    v8f acc = {0.f,0.f,0.f,0.f,0.f,0.f,0.f,0.f};
    for (int kt = 0; kt < ktiles; ++kt) {
      v16h a = load_frag(A,  r0, kt*32, lda, lane);
      v16h b = load_frag(Wt, n0, kt*32, ldw, lane);
      acc = __builtin_amdgcn_wmma_f32_16x16x32_f16(false, a, false, b,
                                                   (short)0, acc, false, false);
    }
    const int n  = n0 + (lane & 15);
    const int mh = (lane & 16) ? 8 : 0;
#pragma unroll
    for (int j = 0; j < 8; ++j)
      Y[(r0 + j + mh) * ldy + n] += acc[j];
  }
}

// X = LayerNorm(X + R) over DD cols, f16 residual
__device__ __forceinline__ void ln_resid16(_Float16* X, const _Float16* R,
                                           const float* g, const float* be, int tid) {
  for (int r = tid; r < MROWS; r += NTHREADS) {
    const int b0 = r * LDP;
    float m = 0.f;
    for (int d = 0; d < DD; ++d) m += (float)X[b0+d] + (float)R[b0+d];
    m *= (1.f/DD);
    float var = 0.f;
    for (int d = 0; d < DD; ++d) {
      float t = (float)X[b0+d] + (float)R[b0+d] - m; var += t*t;
    }
    var *= (1.f/DD);
    const float inv = rsqrtf(var + 1e-5f);
    for (int d = 0; d < DD; ++d) {
      float t = ((float)X[b0+d] + (float)R[b0+d] - m) * inv * g[d] + be[d];
      X[b0+d] = (_Float16)t;
    }
    for (int d = DD; d < LDP; ++d) X[b0+d] = (_Float16)0.f;  // keep K-pad zero
  }
}

// X = LayerNorm(X + Y) over DD cols, f32 residual (stride DPAD)
__device__ __forceinline__ void ln_resid32(_Float16* X, const float* Y,
                                           const float* g, const float* be, int tid) {
  for (int r = tid; r < MROWS; r += NTHREADS) {
    const int b0 = r * LDP;
    const int y0 = r * DPAD;
    float m = 0.f;
    for (int d = 0; d < DD; ++d) m += (float)X[b0+d] + Y[y0+d];
    m *= (1.f/DD);
    float var = 0.f;
    for (int d = 0; d < DD; ++d) {
      float t = (float)X[b0+d] + Y[y0+d] - m; var += t*t;
    }
    var *= (1.f/DD);
    const float inv = rsqrtf(var + 1e-5f);
    for (int d = 0; d < DD; ++d) {
      float t = ((float)X[b0+d] + Y[y0+d] - m) * inv * g[d] + be[d];
      X[b0+d] = (_Float16)t;
    }
    for (int d = DD; d < LDP; ++d) X[b0+d] = (_Float16)0.f;
  }
}

// ---- weight prep: f32 (K x N) -> f16 padded transpose (Npad x Kpad) ----
__global__ void sthd_wprep(const float* __restrict__ W, _Float16* __restrict__ Wt,
                           int K, int N, int Kpad, int Npad) {
  const int idx = blockIdx.x * blockDim.x + threadIdx.x;
  if (idx >= Npad * Kpad) return;
  const int n = idx / Kpad, k = idx % Kpad;
  const float v = (k < K && n < N) ? W[k * N + n] : 0.f;
  Wt[idx] = (_Float16)v;
}

// ---- fully fused STHDformer: one block = 8 sequences, all 4 layers in LDS ----
__global__ void __launch_bounds__(NTHREADS)
sthd_fused(Params P)
{
  extern __shared__ __align__(16) unsigned char smem_raw[];
  _Float16* sX  = (_Float16*)smem_raw;       // current activation (96 x 168)
  _Float16* sHT = sX + BUFH;                 // embedding kept for cross layer
  _Float16* sQ  = sX + 2*BUFH;
  _Float16* sK  = sX + 3*BUFH;
  _Float16* sV  = sX + 4*BUFH;
  _Float16* sH  = sQ;                        // FFN hidden chunk overlays Q
  float*    sY  = (float*)(sX + 3*BUFH);     // FFN f32 accum overlays K+V

  const int tid  = threadIdx.x;
  const int lane = tid & 31;
  const int wid  = tid >> 5;
  const int seq0 = blockIdx.x * SPB;

  // ---- embedding: value proj + tod/dow table lookup + adaptive emb ----
  for (int idx = tid; idx < MROWS*LDP; idx += NTHREADS) {
    const int r = idx / LDP, c = idx % LDP;
    const int s = seq0 + r / TT;
    const int b = s / NNODES, n = s % NNODES;
    const int t = r % TT;
    const int xbase = ((b*TT + t)*NNODES + n) * 3;
    float v = 0.f;
    if (c < 24) {
      v = P.x[xbase] * P.Wi[c] + P.bi[c];
    } else if (c < 48) {
      int ti = (int)(P.x[xbase + 1] * 288.f);
      ti = ti < 0 ? 0 : (ti > 287 ? 287 : ti);
      v = P.todT[ti*24 + (c - 24)];
    } else if (c < 72) {
      int di = (int)P.x[xbase + 2];
      di = di < 0 ? 0 : (di > 6 ? 6 : di);
      v = P.dowT[di*24 + (c - 48)];
    } else if (c < DD) {
      v = P.adp[(t*NNODES + n)*80 + (c - 72)];
    }
    const _Float16 hv = (_Float16)v;
    sHT[idx] = hv;
    sX[idx]  = hv;
  }
  __syncthreads();

  const float scale = 0.16222142f;  // 1/sqrt(38)

  for (int l = 0; l < 4; ++l) {     // 3 self layers + 1 cross layer
    const _Float16* Wl = P.W + (size_t)l * LAYER_WS;
    const LayerP Lp = P.L[l];
    const _Float16* kvsrc = (l == 3) ? sHT : sX;

    // Q/K/V projections (WMMA)
    gemm_f16out(sX,    LDP, Wl + WQT_OFF, DPAD, Lp.bq, DD, sQ, LDP, MT, NT_D, KT_D, false, wid, lane);
    gemm_f16out(kvsrc, LDP, Wl + WKT_OFF, DPAD, Lp.bk, DD, sK, LDP, MT, NT_D, KT_D, false, wid, lane);
    gemm_f16out(kvsrc, LDP, Wl + WVT_OFF, DPAD, Lp.bv, DD, sV, LDP, MT, NT_D, KT_D, false, wid, lane);
    __syncthreads();

    // attention over T=12 (tiny; f32 VALU), result written over own Q slice
    for (int task = tid; task < SPB*NHEAD*TT; task += NTHREADS) {
      const int qt = task % TT;
      const int h  = (task / TT) % NHEAD;
      const int sl = task / (TT*NHEAD);
      _Float16* Qr = sQ + (sl*TT + qt)*LDP + h*HDIM;
      float sc[TT];
      float mx = -1e30f;
      for (int kk = 0; kk < TT; ++kk) {
        const _Float16* Kr = sK + (sl*TT + kk)*LDP + h*HDIM;
        float d = 0.f;
        for (int i = 0; i < HDIM; ++i) d += (float)Qr[i] * (float)Kr[i];
        d *= scale;
        sc[kk] = d;
        mx = fmaxf(mx, d);
      }
      float sum = 0.f;
      for (int kk = 0; kk < TT; ++kk) { sc[kk] = __expf(sc[kk] - mx); sum += sc[kk]; }
      const float inv = 1.f / sum;
      float o[HDIM];
      for (int i = 0; i < HDIM; ++i) o[i] = 0.f;
      for (int kk = 0; kk < TT; ++kk) {
        const float w = sc[kk] * inv;
        const _Float16* Vr = sV + (sl*TT + kk)*LDP + h*HDIM;
        for (int i = 0; i < HDIM; ++i) o[i] += w * (float)Vr[i];
      }
      for (int i = 0; i < HDIM; ++i) Qr[i] = (_Float16)o[i];
    }
    __syncthreads();

    // output projection of attention, then residual + LN1
    gemm_f16out(sQ, LDP, Wl + WOT_OFF, DPAD, Lp.bo, DD, sK, LDP, MT, NT_D, KT_D, false, wid, lane);
    __syncthreads();
    ln_resid16(sX, sK, Lp.g1, Lp.be1, tid);
    __syncthreads();

    // FFN: chunk FF=512 in 4 slices of 128; Y accumulates in f32 LDS
    for (int idx = tid; idx < MROWS*DPAD; idx += NTHREADS) {
      const int n = idx % DPAD;
      sY[idx] = (n < DD) ? Lp.b2[n] : 0.f;
    }
    __syncthreads();
    for (int c = 0; c < FFD/FFC; ++c) {
      gemm_f16out(sX, LDP, Wl + W1T_OFF + c*FFC*DPAD, DPAD, Lp.b1 + c*FFC, FFC,
                  sH, LDH, MT, FFC/16, KT_D, true, wid, lane);
      __syncthreads();
      gemm_f32acc(sH, LDH, Wl + W2T_OFF + c*FFC, FFD, sY, DPAD,
                  MT, NT_D, FFC/32, wid, lane);
      __syncthreads();
    }
    ln_resid32(sX, sY, Lp.g2, Lp.be2, tid);
    __syncthreads();
  }

  // ---- output head: (T*D=1824) -> 12 per sequence ----
  for (int task = tid; task < SPB*OUTS; task += NTHREADS) {
    const int ot = task % OUTS;
    const int sl = task / OUTS;
    const int s  = seq0 + sl;
    const int b  = s / NNODES, n = s % NNODES;
    float acc = P.bout[ot];
    for (int t = 0; t < TT; ++t) {
      const _Float16* row = sX + (sl*TT + t)*LDP;
      const float* wr = P.Wout + t*DD*OUTS + ot;
      for (int d = 0; d < DD; ++d)
        acc += (float)row[d] * wr[d*OUTS];
    }
    P.out[(b*OUTS + ot)*NNODES + n] = acc;
  }
}

extern "C" void kernel_launch(void* const* d_in, const int* in_sizes, int n_in,
                              void* d_out, int out_size, void* d_ws, size_t ws_size,
                              hipStream_t stream) {
  (void)in_sizes; (void)n_in; (void)out_size; (void)ws_size;
  auto F = [&](int i) { return (const float*)d_in[i]; };
  _Float16* wsW = (_Float16*)d_ws;
  const int bases[4] = {6, 22, 38, 54};   // layer0..2, cross

  // weight prep: transpose+pad to f16 W^T (every call; deterministic, ~1M elts)
  auto tr = [&](const float* W, _Float16* Wt, int K, int N, int Kpad, int Npad) {
    const int total = Npad * Kpad;
    sthd_wprep<<<(total + 255)/256, 256, 0, stream>>>(W, Wt, K, N, Kpad, Npad);
  };
  for (int l = 0; l < 4; ++l) {
    const int b0 = bases[l];
    _Float16* wl = wsW + (size_t)l * LAYER_WS;
    tr(F(b0+0),  wl + WQT_OFF, DD,  DD,  DPAD, DPAD);
    tr(F(b0+2),  wl + WKT_OFF, DD,  DD,  DPAD, DPAD);
    tr(F(b0+4),  wl + WVT_OFF, DD,  DD,  DPAD, DPAD);
    tr(F(b0+6),  wl + WOT_OFF, DD,  DD,  DPAD, DPAD);
    tr(F(b0+8),  wl + W1T_OFF, DD,  FFD, DPAD, FFD);
    tr(F(b0+10), wl + W2T_OFF, FFD, DD,  FFD,  DPAD);
  }

  Params P;
  P.x = F(0); P.Wi = F(1); P.bi = F(2); P.todT = F(3); P.dowT = F(4); P.adp = F(5);
  P.Wout = F(70); P.bout = F(71);
  P.W = (const _Float16*)d_ws;
  P.out = (float*)d_out;
  for (int l = 0; l < 4; ++l) {
    const int b0 = bases[l];
    P.L[l].bq  = F(b0+1);  P.L[l].bk  = F(b0+3);  P.L[l].bv = F(b0+5);
    P.L[l].bo  = F(b0+7);  P.L[l].b1  = F(b0+9);  P.L[l].b2 = F(b0+11);
    P.L[l].g1  = F(b0+12); P.L[l].be1 = F(b0+13);
    P.L[l].g2  = F(b0+14); P.L[l].be2 = F(b0+15);
  }

  hipFuncSetAttribute((const void*)sthd_fused,
                      hipFuncAttributeMaxDynamicSharedMemorySize, SMEM_BYTES);
  const int nblocks = (BB * NNODES) / SPB;   // 2864
  sthd_fused<<<nblocks, NTHREADS, SMEM_BYTES, stream>>>(P);
}
